// Mamba2SSD_17248588660840
// MI455X (gfx1250) — matrix operand order repaired and checked by reading, not compiled, not run on hardware
//
#include <hip/hip_runtime.h>
#include <hip/hip_bf16.h>

// ---------------------------------------------------------------------------
// Mamba2 SSD block for MI455X (gfx1250, wave32, WMMA).
// All matmuls run through v_wmma_f32_16x16x32_bf16 (f32 accumulate);
// fp32->bf16 staging uses v_cvt_pk_bf16_f32.
// ---------------------------------------------------------------------------

namespace {
constexpr int kB   = 2;
constexpr int kS   = 2048;
constexpr int kDM  = 2048;
constexpr int kDI  = 4096;   // D_INNER
constexpr int kH   = 64;     // heads
constexpr int kP   = 64;     // headdim
constexpr int kN   = 64;     // d_state
constexpr int kNC  = 32;     // chunks
constexpr int kCH  = 64;     // chunk length
constexpr int kX   = 192;    // 2*d_state + heads
}

typedef __attribute__((ext_vector_type(16))) __bf16 v16bf;
typedef __attribute__((ext_vector_type(8)))  float  v8f;

// Pack two fp32 -> packed bf16 (RNE). One v_cvt_pk_bf16_f32 on gfx1250.
__device__ __forceinline__ unsigned pk2(float a, float b) {
#if defined(__HIP_DEVICE_COMPILE__)
  unsigned r;
  asm("v_cvt_pk_bf16_f32 %0, %1, %2" : "=v"(r) : "v"(a), "v"(b));
  return r;
#else
  union { float f; unsigned u; } va, vb; va.f = a; vb.f = b;
  unsigned ra = (va.u + 0x7FFFu + ((va.u >> 16) & 1u)) >> 16;
  unsigned rb = (vb.u + 0x7FFFu + ((vb.u >> 16) & 1u)) & 0xFFFF0000u;
  return ra | rb;
#endif
}
__device__ __forceinline__ __bf16 f2bf(float f) {
  unsigned r = pk2(f, f);
  unsigned short s = (unsigned short)r;
  __bf16 o; __builtin_memcpy(&o, &s, 2); return o;
}
__device__ __forceinline__ float bf2f(__bf16 h) {
  unsigned short s; __builtin_memcpy(&s, &h, 2);
  union { unsigned u; float f; } v; v.u = ((unsigned)s) << 16; return v.f;
}
// 8 floats -> 8 bf16, one b128 LDS store (dst must be 16B aligned)
__device__ __forceinline__ void store_pk8(__bf16* dst, float4 lo, float4 hi) {
  uint4 u;
  u.x = pk2(lo.x, lo.y); u.y = pk2(lo.z, lo.w);
  u.z = pk2(hi.x, hi.y); u.w = pk2(hi.z, hi.w);
  *(uint4*)dst = u;
}
// 4 floats -> 4 bf16, one b64 LDS store (dst must be 8B aligned)
__device__ __forceinline__ void store_pk4(__bf16* dst, float4 v) {
  uint2 u;
  u.x = pk2(v.x, v.y); u.y = pk2(v.z, v.w);
  *(uint2*)dst = u;
}
__device__ __forceinline__ v8f wmma_bf16(v16bf a, v16bf b, v8f c) {
#if defined(__HIP_DEVICE_COMPILE__)
  return __builtin_amdgcn_wmma_f32_16x16x32_bf16(
      false, a, false, b, (short)0, c, false, false);
#else
  return c;
#endif
}
__device__ __forceinline__ float silu(float x) { return x / (1.f + __expf(-x)); }

// ---------------------------------------------------------------------------
// GEMM: C[M,N] = A[M,K] @ W[N,K]^T   (fp32 row-major in HBM, bf16 WMMA)
// Workgroup 256 threads = 8 waves; tile 128(M) x 64(N); K-step 32.
// Wave w owns rows [16w,16w+16) x all 64 cols: 4 WMMAs per K-step.
// ---------------------------------------------------------------------------
__global__ __launch_bounds__(256) void gemm_wmma_kernel(
    const float* __restrict__ A, const float* __restrict__ W,
    float* __restrict__ C, int M, int N, int K)
{
  __shared__ __align__(16) __bf16 As[128][40];   // row stride 80B (16B mult)
  __shared__ __align__(16) __bf16 Ws[64][40];
  const int bm = blockIdx.y, bn = blockIdx.x;
  const int tid  = threadIdx.x;
  const int lane = tid & 31, wave = tid >> 5;
  const int ml = lane & 15, hi = lane >> 4;
  const int lr = tid >> 2;            // staging row 0..63
  const int c8 = (tid & 3) * 8;       // staging col chunk

  v8f acc[4];
  const v8f vzero = {0.f, 0.f, 0.f, 0.f, 0.f, 0.f, 0.f, 0.f};
  #pragma unroll
  for (int t = 0; t < 4; ++t) acc[t] = vzero;

  for (int k0 = 0; k0 < K; k0 += 32) {
    // Stage A (128x32) and W (64x32) tiles as bf16 via b128 loads/stores.
    {
      const float* s0 = A + (size_t)(bm * 128 + lr) * K + k0 + c8;
      const float* s1 = A + (size_t)(bm * 128 + 64 + lr) * K + k0 + c8;
      const float* sw = W + (size_t)(bn * 64 + lr) * K + k0 + c8;
      float4 a0 = *(const float4*)s0, a1 = *(const float4*)(s0 + 4);
      float4 a2 = *(const float4*)s1, a3 = *(const float4*)(s1 + 4);
      float4 w0 = *(const float4*)sw, w1 = *(const float4*)(sw + 4);
      store_pk8(&As[lr][c8],      a0, a1);
      store_pk8(&As[64 + lr][c8], a2, a3);
      store_pk8(&Ws[lr][c8],      w0, w1);
    }
    if (k0 + 32 < K) {  // global_prefetch hint for the next K tile
      __builtin_prefetch(A + (size_t)(bm * 128 + lr) * K + k0 + 32 + c8, 0, 1);
      __builtin_prefetch(W + (size_t)(bn * 64 + lr) * K + k0 + 32 + c8, 0, 1);
    }
    __syncthreads();

    // A fragment (16x32): lanes0-15 K0..7|K16..23, lanes16-31 +8
    v16bf a;
    const int ar = wave * 16 + ml;
    #pragma unroll
    for (int j = 0; j < 8; ++j) a[j]     = As[ar][j + 8 * hi];
    #pragma unroll
    for (int j = 0; j < 8; ++j) a[8 + j] = As[ar][16 + j + 8 * hi];
    // 4 B fragments (32x16 each): lanes0-15 K0..7|K16..23, lanes16-31 +8
    #pragma unroll
    for (int t = 0; t < 4; ++t) {
      v16bf bb;
      const int br = t * 16 + ml;
      #pragma unroll
      for (int j = 0; j < 8; ++j) bb[j]     = Ws[br][j + 8 * hi];
      #pragma unroll
      for (int j = 0; j < 8; ++j) bb[8 + j] = Ws[br][16 + j + 8 * hi];
      acc[t] = wmma_bf16(a, bb, acc[t]);
    }
    __syncthreads();
  }
  const int row0 = bm * 128 + wave * 16 + 8 * hi;
  #pragma unroll
  for (int t = 0; t < 4; ++t) {
    const int col = bn * 64 + t * 16 + ml;
    #pragma unroll
    for (int v = 0; v < 8; ++v)
      C[(size_t)(row0 + v) * N + col] = acc[t][v];
  }
}

// ---------------------------------------------------------------------------
// Depthwise causal conv(4) + SiLU, 4 channels per thread (b128 traffic).
// ---------------------------------------------------------------------------
__global__ __launch_bounds__(256) void conv_silu_kernel(
    const float* __restrict__ xz, const float* __restrict__ conv_w,
    const float* __restrict__ conv_b, float* __restrict__ xconv)
{
  const int idx = blockIdx.x * 256 + threadIdx.x;     // over B*S*DI/4
  const int c4  = (idx & (kDI / 4 - 1)) * 4;
  const int bs  = idx >> 10;                          // row = b*S + s
  const int s   = bs & (kS - 1);
  float4 acc = *(const float4*)(conv_b + c4);
  #pragma unroll
  for (int j = 0; j < 4; ++j) {
    int sp = s - 3 + j;
    if (sp >= 0) {
      float4 xv = *(const float4*)(xz + (size_t)(bs + sp - s) * (2 * kDI) + c4);
      acc.x += xv.x * conv_w[(c4 + 0) * 4 + j];
      acc.y += xv.y * conv_w[(c4 + 1) * 4 + j];
      acc.z += xv.z * conv_w[(c4 + 2) * 4 + j];
      acc.w += xv.w * conv_w[(c4 + 3) * 4 + j];
    }
  }
  float4 o;
  o.x = silu(acc.x); o.y = silu(acc.y); o.z = silu(acc.z); o.w = silu(acc.w);
  *(float4*)(xconv + (size_t)idx * 4) = o;
}

// ---------------------------------------------------------------------------
// A_dt[row,h] = -exp(A_log[h]) * softplus(x_ssm[row, 128+h])
// ---------------------------------------------------------------------------
__global__ __launch_bounds__(256) void adt_kernel(
    const float* __restrict__ xssm, const float* __restrict__ A_log,
    float* __restrict__ adt)
{
  const int idx = blockIdx.x * 256 + threadIdx.x;     // over B*S*H
  const int h   = idx & (kH - 1);
  const int row = idx >> 6;
  float d  = xssm[(size_t)row * kX + 128 + h];
  float sp = (d > 20.f) ? d : log1pf(__expf(d));
  adt[idx] = -__expf(A_log[h]) * sp;
}

// ---------------------------------------------------------------------------
// Per-chunk inclusive cumsum of A_dt (one thread per (b, chunk, head)).
// ---------------------------------------------------------------------------
__global__ __launch_bounds__(256) void cumsum_kernel(
    const float* __restrict__ adt, float* __restrict__ acum)
{
  const int idx = blockIdx.x * 256 + threadIdx.x;     // over B*NC*H
  const int h = idx & (kH - 1);
  const int c = (idx >> 6) & (kNC - 1);
  const int b = idx >> 11;
  size_t base = ((size_t)(b * kS + c * kCH)) * kH + h;
  float run = 0.f;
  for (int i = 0; i < kCH; ++i) {
    run += adt[base + (size_t)i * kH];
    acum[base + (size_t)i * kH] = run;
  }
}

// ---------------------------------------------------------------------------
// SSD intra-chunk: one workgroup per (b, chunk, head). 8 waves.
// ---------------------------------------------------------------------------
__global__ __launch_bounds__(256) void ssd_intra_kernel(
    const float* __restrict__ xssm, const float* __restrict__ xconv,
    const float* __restrict__ acum, float* __restrict__ y,
    float* __restrict__ states)
{
  const int blk = blockIdx.x;                 // (b*NC + c)*H + h
  const int h = blk & (kH - 1);
  const int c = (blk >> 6) & (kNC - 1);
  const int b = blk >> 11;

  __shared__ __align__(16) __bf16 Cs[kCH][kN + 8];   // row stride 144B
  __shared__ __align__(16) __bf16 Bs[kCH][kN + 8];
  __shared__ __align__(16) __bf16 Xs[kCH][kP + 8];
  __shared__ __align__(16) __bf16 Ls[kCH][kCH + 8];
  __shared__ __align__(16) __bf16 Bd[kCH][kN + 8];   // B * decay_to_end
  __shared__ float AcS[kCH];

  const int tid = threadIdx.x;
  const size_t rowbase = (size_t)(b * kS + c * kCH);
  #pragma unroll
  for (int it = 0; it < 4; ++it) {
    int i = it * 256 + tid;
    int r = i >> 4, c4 = (i & 15) * 4;
    const float* xs = xssm + (rowbase + r) * kX;
    store_pk4(&Bs[r][c4], *(const float4*)(xs + c4));
    store_pk4(&Cs[r][c4], *(const float4*)(xs + 64 + c4));
    store_pk4(&Xs[r][c4],
              *(const float4*)(xconv + (rowbase + r) * kDI + h * kP + c4));
  }
  if (tid < kCH) AcS[tid] = acum[(rowbase + tid) * kH + h];
  __syncthreads();

  // ---- Pre-pass: Bd[k][n] = Bs[k][n] * exp(Acum_last - Acum[k])
  const float alast = AcS[kCH - 1];
  #pragma unroll
  for (int it = 0; it < 4; ++it) {
    int i = it * 256 + tid;
    int r = i >> 4, c4 = (i & 15) * 4;
    const float d = __expf(alast - AcS[r]);
    float4 v;
    v.x = bf2f(Bs[r][c4 + 0]) * d;
    v.y = bf2f(Bs[r][c4 + 1]) * d;
    v.z = bf2f(Bs[r][c4 + 2]) * d;
    v.w = bf2f(Bs[r][c4 + 3]) * d;
    store_pk4(&Bd[r][c4], v);
  }
  __syncthreads();

  const int lane = tid & 31, wave = tid >> 5;
  const int ml = lane & 15, hi = lane >> 4;

  // ---- Stage 1: CB = Cc @ Bc^T with causal decay mask -> Ls
  #pragma unroll
  for (int rep = 0; rep < 2; ++rep) {
    const int tile = wave + rep * 8;          // 16 tiles of 4x4 grid
    const int ti = tile >> 2, tj = tile & 3;
    v8f acc = {};
    #pragma unroll
    for (int k0 = 0; k0 < kN; k0 += 32) {
      v16bf a, bb;
      const int ar = ti * 16 + ml;
      #pragma unroll
      for (int j = 0; j < 8; ++j) a[j]     = Cs[ar][k0 + j + 8 * hi];
      #pragma unroll
      for (int j = 0; j < 8; ++j) a[8 + j] = Cs[ar][k0 + 16 + j + 8 * hi];
      const int br = tj * 16 + ml;
      #pragma unroll
      for (int j = 0; j < 8; ++j) bb[j]     = Bs[br][k0 + j + 8 * hi];
      #pragma unroll
      for (int j = 0; j < 8; ++j) bb[8 + j] = Bs[br][k0 + 16 + j + 8 * hi];
      acc = wmma_bf16(a, bb, acc);
    }
    const int col = tj * 16 + ml;
    #pragma unroll
    for (int v = 0; v < 8; ++v) {
      const int row = ti * 16 + 8 * hi + v;
      float val = (row >= col) ? acc[v] * __expf(AcS[row] - AcS[col]) : 0.f;
      Ls[row][col] = f2bf(val);
    }
  }
  __syncthreads();

  // ---- Stage 2: y_intra = Ls @ Xs -> global y  [t x p]
  #pragma unroll
  for (int rep = 0; rep < 2; ++rep) {
    const int tile = wave + rep * 8;
    const int ti = tile >> 2, tj = tile & 3;
    v8f acc = {};
    #pragma unroll
    for (int k0 = 0; k0 < kCH; k0 += 32) {
      v16bf a, bb;
      const int ar = ti * 16 + ml;
      #pragma unroll
      for (int j = 0; j < 8; ++j) a[j]     = Ls[ar][k0 + j + 8 * hi];
      #pragma unroll
      for (int j = 0; j < 8; ++j) a[8 + j] = Ls[ar][k0 + 16 + j + 8 * hi];
      const int pc = tj * 16 + ml;
      #pragma unroll
      for (int j = 0; j < 8; ++j) bb[j]     = Xs[k0 + j + 8 * hi][pc];
      #pragma unroll
      for (int j = 0; j < 8; ++j) bb[8 + j] = Xs[k0 + 16 + j + 8 * hi][pc];
      acc = wmma_bf16(a, bb, acc);
    }
    const int col = tj * 16 + ml;
    #pragma unroll
    for (int v = 0; v < 8; ++v) {
      const int row = ti * 16 + 8 * hi + v;
      y[(rowbase + row) * kDI + h * kP + col] = acc[v];
    }
  }

  // ---- Stage 3: states = Bd^T @ Xs  [n x p]  (A[n][k] = Bd[k][n])
  #pragma unroll
  for (int rep = 0; rep < 2; ++rep) {
    const int tile = wave + rep * 8;
    const int ti = tile >> 2, tj = tile & 3;
    v8f acc = {};
    #pragma unroll
    for (int k0 = 0; k0 < kCH; k0 += 32) {
      v16bf a, bb;
      const int nr = ti * 16 + ml;
      #pragma unroll
      for (int j = 0; j < 8; ++j) a[j]     = Bd[k0 + j + 8 * hi][nr];
      #pragma unroll
      for (int j = 0; j < 8; ++j) a[8 + j] = Bd[k0 + 16 + j + 8 * hi][nr];
      const int pc = tj * 16 + ml;
      #pragma unroll
      for (int j = 0; j < 8; ++j) bb[j]     = Xs[k0 + j + 8 * hi][pc];
      #pragma unroll
      for (int j = 0; j < 8; ++j) bb[8 + j] = Xs[k0 + 16 + j + 8 * hi][pc];
      acc = wmma_bf16(a, bb, acc);
    }
    const int col = tj * 16 + ml;
    size_t sb = (size_t)blk * (kN * kP);
    #pragma unroll
    for (int v = 0; v < 8; ++v) {
      const int nrow = ti * 16 + 8 * hi + v;
      states[sb + (size_t)nrow * kP + col] = acc[v];
    }
  }
}

// ---------------------------------------------------------------------------
// Inter-chunk sequential scan: one workgroup per (b, head); float4 per step.
// ---------------------------------------------------------------------------
__global__ __launch_bounds__(256) void ssd_scan_kernel(
    const float* __restrict__ states, const float* __restrict__ acum,
    float* __restrict__ inits)
{
  const int bh = blockIdx.x;
  const int b = bh >> 6, h = bh & (kH - 1);
  const int t = threadIdx.x;
  float4 carry[4];
  #pragma unroll
  for (int e = 0; e < 4; ++e) carry[e] = make_float4(0.f, 0.f, 0.f, 0.f);
  for (int c = 0; c < kNC; ++c) {
    const float dec =
        __expf(acum[((size_t)(b * kS + c * kCH + kCH - 1)) * kH + h]);
    const size_t base = ((size_t)((b * kNC + c) * kH + h)) * (kN * kP);
    #pragma unroll
    for (int e = 0; e < 4; ++e) {
      size_t off = base + (size_t)e * 1024 + t * 4;
      *(float4*)(inits + off) = carry[e];
      float4 sv = *(const float4*)(states + off);
      carry[e].x = dec * carry[e].x + sv.x;
      carry[e].y = dec * carry[e].y + sv.y;
      carry[e].z = dec * carry[e].z + sv.z;
      carry[e].w = dec * carry[e].w + sv.w;
    }
  }
}

// ---------------------------------------------------------------------------
// y += decay_excl[t] * (Cc @ state0)  -- one workgroup per (b, chunk, head)
// ---------------------------------------------------------------------------
__global__ __launch_bounds__(256) void ssd_inter_kernel(
    const float* __restrict__ xssm, const float* __restrict__ inits,
    const float* __restrict__ acum, const float* __restrict__ adt,
    float* __restrict__ y)
{
  const int blk = blockIdx.x;
  const int h = blk & (kH - 1);
  const int c = (blk >> 6) & (kNC - 1);
  const int b = blk >> 11;

  __shared__ __align__(16) __bf16 Cs[kCH][kN + 8];
  __shared__ __align__(16) __bf16 Is[kN][kP + 8];
  __shared__ float DeS[kCH];

  const int tid = threadIdx.x;
  const size_t rowbase = (size_t)(b * kS + c * kCH);
  const size_t sb = (size_t)blk * (kN * kP);
  #pragma unroll
  for (int it = 0; it < 4; ++it) {
    int i = it * 256 + tid;
    int r = i >> 4, c4 = (i & 15) * 4;
    store_pk4(&Cs[r][c4], *(const float4*)(xssm + (rowbase + r) * kX + 64 + c4));
    store_pk4(&Is[r][c4], *(const float4*)(inits + sb + (size_t)r * kP + c4));
  }
  if (tid < kCH) {
    size_t off = (rowbase + tid) * kH + h;
    DeS[tid] = __expf(acum[off] - adt[off]);
  }
  __syncthreads();

  const int lane = tid & 31, wave = tid >> 5;
  const int ml = lane & 15, hi = lane >> 4;
  #pragma unroll
  for (int rep = 0; rep < 2; ++rep) {
    const int tile = wave + rep * 8;
    const int ti = tile >> 2, tj = tile & 3;
    v8f acc = {};
    #pragma unroll
    for (int k0 = 0; k0 < kN; k0 += 32) {
      v16bf a, bb;
      const int ar = ti * 16 + ml;
      #pragma unroll
      for (int j = 0; j < 8; ++j) a[j]     = Cs[ar][k0 + j + 8 * hi];
      #pragma unroll
      for (int j = 0; j < 8; ++j) a[8 + j] = Cs[ar][k0 + 16 + j + 8 * hi];
      const int pc = tj * 16 + ml;
      #pragma unroll
      for (int j = 0; j < 8; ++j) bb[j]     = Is[k0 + j + 8 * hi][pc];
      #pragma unroll
      for (int j = 0; j < 8; ++j) bb[8 + j] = Is[k0 + 16 + j + 8 * hi][pc];
      acc = wmma_bf16(a, bb, acc);
    }
    const int col = tj * 16 + ml;
    #pragma unroll
    for (int v = 0; v < 8; ++v) {
      const int row = ti * 16 + 8 * hi + v;
      size_t off = (rowbase + row) * kDI + h * kP + col;
      y[off] += DeS[row] * acc[v];
    }
  }
}

// ---------------------------------------------------------------------------
// Fused: y += x_conv*D ; y *= silu(z) ; RMSNorm(y)*norm_w   (in place, b128)
// ---------------------------------------------------------------------------
__global__ __launch_bounds__(256) void finalize_kernel(
    const float* __restrict__ xz, const float* __restrict__ xconv,
    const float* __restrict__ D_par, const float* __restrict__ norm_w,
    float* __restrict__ y)
{
  const int row = blockIdx.x;                   // b*S + s
  const size_t base  = (size_t)row * kDI;
  const size_t zbase = (size_t)row * (2 * kDI) + kDI;
  __shared__ float red[8];
  float4 vals[4];
  float ss = 0.f;
  #pragma unroll
  for (int i = 0; i < 4; ++i) {
    const int c4 = (i * 256 + threadIdx.x) * 4;
    float4 yv = *(const float4*)(y + base + c4);
    float4 xc = *(const float4*)(xconv + base + c4);
    float4 zv = *(const float4*)(xz + zbase + c4);
    const float dp = D_par[c4 >> 6];            // same head for all 4 lanes
    float4 v;
    v.x = (yv.x + xc.x * dp) * silu(zv.x);
    v.y = (yv.y + xc.y * dp) * silu(zv.y);
    v.z = (yv.z + xc.z * dp) * silu(zv.z);
    v.w = (yv.w + xc.w * dp) * silu(zv.w);
    vals[i] = v;
    ss += v.x * v.x + v.y * v.y + v.z * v.z + v.w * v.w;
  }
  #pragma unroll
  for (int off = 16; off > 0; off >>= 1) ss += __shfl_down(ss, off, 32);
  if ((threadIdx.x & 31) == 0) red[threadIdx.x >> 5] = ss;
  __syncthreads();
  if (threadIdx.x == 0) {
    float t = 0.f;
    #pragma unroll
    for (int w = 0; w < 8; ++w) t += red[w];
    red[0] = t;
  }
  __syncthreads();
  const float rinv = rsqrtf(red[0] / (float)kDI + 1e-6f);
  #pragma unroll
  for (int i = 0; i < 4; ++i) {
    const int c4 = (i * 256 + threadIdx.x) * 4;
    float4 nw = *(const float4*)(norm_w + c4);
    float4 v = vals[i];
    v.x *= rinv * nw.x; v.y *= rinv * nw.y;
    v.z *= rinv * nw.z; v.w *= rinv * nw.w;
    *(float4*)(y + base + c4) = v;
  }
}

// ---------------------------------------------------------------------------
extern "C" void kernel_launch(void* const* d_in, const int* in_sizes, int n_in,
                              void* d_out, int out_size, void* d_ws, size_t ws_size,
                              hipStream_t stream) {
  const float* x      = (const float*)d_in[0];
  const float* w_in   = (const float*)d_in[1];
  const float* conv_w = (const float*)d_in[2];
  const float* conv_b = (const float*)d_in[3];
  const float* w_x    = (const float*)d_in[4];
  const float* A_log  = (const float*)d_in[5];
  const float* D_par  = (const float*)d_in[6];
  const float* w_out  = (const float*)d_in[7];
  const float* norm_w = (const float*)d_in[8];
  float* out = (float*)d_out;

  float* ws = (float*)d_ws;
  float* xz     = ws; ws += (size_t)kB * kS * 2 * kDI;        // 134 MB
  float* xconv  = ws; ws += (size_t)kB * kS * kDI;            //  67 MB
  float* xssm   = ws; ws += (size_t)kB * kS * kX;             //   3 MB
  float* adt    = ws; ws += (size_t)kB * kS * kH;             //   1 MB
  float* acum   = ws; ws += (size_t)kB * kS * kH;             //   1 MB
  float* ybuf   = ws; ws += (size_t)kB * kS * kDI;            //  67 MB
  float* states = ws; ws += (size_t)kB * kNC * kH * kN * kP;  //  67 MB
  float* inits  = ws; ws += (size_t)kB * kNC * kH * kN * kP;  //  67 MB

  const dim3 blk(256);
  const int M = kB * kS;

  // 1. xz = x @ w_in^T                 [4096 x 8192, K=2048]
  gemm_wmma_kernel<<<dim3((2 * kDI) / 64, M / 128), blk, 0, stream>>>(
      x, w_in, xz, M, 2 * kDI, kDM);
  // 2. depthwise conv + SiLU
  conv_silu_kernel<<<(kB * kS * kDI / 4) / 256, blk, 0, stream>>>(
      xz, conv_w, conv_b, xconv);
  // 3. x_ssm = x_conv @ w_x^T          [4096 x 192, K=4096]
  gemm_wmma_kernel<<<dim3(kX / 64, M / 128), blk, 0, stream>>>(
      xconv, w_x, xssm, M, kX, kDI);
  // 4. A_dt
  adt_kernel<<<(kB * kS * kH) / 256, blk, 0, stream>>>(xssm, A_log, adt);
  // 5. per-chunk cumsum
  cumsum_kernel<<<(kB * kNC * kH) / 256, blk, 0, stream>>>(adt, acum);
  // 6. intra-chunk SSD (WMMA) + chunk states
  ssd_intra_kernel<<<kB * kNC * kH, blk, 0, stream>>>(
      xssm, xconv, acum, ybuf, states);
  // 7. inter-chunk recurrence
  ssd_scan_kernel<<<kB * kH, blk, 0, stream>>>(states, acum, inits);
  // 8. inter-chunk contribution (WMMA)
  ssd_inter_kernel<<<kB * kNC * kH, blk, 0, stream>>>(
      xssm, inits, acum, adt, ybuf);
  // 9. D residual + gate + RMSNorm (in place)
  finalize_kernel<<<kB * kS, blk, 0, stream>>>(xz, xconv, D_par, norm_w, ybuf);
  // 10. out = y @ w_out^T              [4096 x 2048, K=4096]
  gemm_wmma_kernel<<<dim3(kDM / 64, M / 128), blk, 0, stream>>>(
      ybuf, w_out, out, M, kDM, kDI);
}
